// Attention_69234872811851
// MI455X (gfx1250) — compile-verified, matrix operand order checked by reading, not run on hardware
//
#include <hip/hip_runtime.h>
#include <hip/hip_bf16.h>

typedef __bf16 bf16;
typedef __attribute__((ext_vector_type(8)))  __bf16 v8bf;
typedef __attribute__((ext_vector_type(16))) __bf16 v16bf;
typedef __attribute__((ext_vector_type(8)))  float  v8f;

#define B_   8
#define N_   1024
#define C_   768
#define H_   12
#define D_   64
#define M_   (B_ * N_)      // 8192 rows
#define SCALE_ 0.125f       // D^-0.5

// ---------------------------------------------------------------------------
// Fragment loader shared by A (row-major, row = lane&15) and B (stored
// transposed so the column is contiguous, col = lane&15).
// CDNA5 16-bit 16x32 A layout: lanes 0-15 hold K 0..7 / 16..23 of row M=lane,
// lanes 16-31 hold K 8..15 / 24..31 of row M=lane-16.  klo = (lane&16)?8:0.
// ---------------------------------------------------------------------------
__device__ __forceinline__ v16bf load_frag(const bf16* p, int klo) {
    v8bf lo = *(const v8bf*)(p + klo);
    v8bf hi = *(const v8bf*)(p + klo + 16);
    v16bf r;
#pragma unroll
    for (int i = 0; i < 8; ++i) { r[i] = lo[i]; r[i + 8] = hi[i]; }
    return r;
}

#define WMMA_BF16(A, B, C) \
    __builtin_amdgcn_wmma_f32_16x16x32_bf16(false, (A), false, (B), (short)0, (C), false, false)

// ---------------------------------------------------------------------------
// fp32 -> bf16 elementwise convert
// ---------------------------------------------------------------------------
__global__ void cvt_f32_bf16(const float* __restrict__ in, bf16* __restrict__ out, int n) {
    int i = blockIdx.x * blockDim.x + threadIdx.x;
    if (i < n) out[i] = (bf16)in[i];
}

// fp32 [K][Nn] -> bf16 [Nn][K] (transpose so GEMM B-operand is K-contiguous)
__global__ void transpose_cvt(const float* __restrict__ in, bf16* __restrict__ out,
                              int K, int Nn) {
    int idx = blockIdx.x * blockDim.x + threadIdx.x;
    if (idx >= K * Nn) return;
    int k = idx / Nn, n = idx - k * Nn;
    out[n * K + k] = (bf16)in[idx];
}

// ---------------------------------------------------------------------------
// QKV GEMM: X[8192][768] x Wt[2304][768]^T  (+bias) -> Q/K (B,H,N,D), Vt (B,H,D,N)
// One wave per 32x32 tile: 2 A-frags x 2 B-frags -> 4 WMMA per 32-wide K step.
// ---------------------------------------------------------------------------
__global__ void __launch_bounds__(128)
qkv_gemm(const bf16* __restrict__ X, const bf16* __restrict__ Wt,
         const float* __restrict__ bias,
         bf16* __restrict__ Qb, bf16* __restrict__ Kb, bf16* __restrict__ Vt) {
    const int lane = threadIdx.x & 31;
    const int wave = blockIdx.x * (blockDim.x >> 5) + (threadIdx.x >> 5);
    const int NT = (3 * C_) / 32;           // 72 N-tiles
    const int mt = wave / NT;               // 0..255
    const int nt = wave - mt * NT;
    const int klo  = (lane & 16) ? 8 : 0;
    const int rsel = lane & 15;

    const bf16* a0 = X  + (size_t)(mt * 32 + rsel) * C_;
    const bf16* a1 = a0 + (size_t)16 * C_;
    const bf16* b0 = Wt + (size_t)(nt * 32 + rsel) * C_;
    const bf16* b1 = b0 + (size_t)16 * C_;

    v8f acc00 = {}, acc01 = {}, acc10 = {}, acc11 = {};
#pragma unroll 4
    for (int k0 = 0; k0 < C_; k0 += 32) {
        v16bf fa0 = load_frag(a0 + k0, klo);
        v16bf fa1 = load_frag(a1 + k0, klo);
        v16bf fb0 = load_frag(b0 + k0, klo);
        v16bf fb1 = load_frag(b1 + k0, klo);
        acc00 = WMMA_BF16(fa0, fb0, acc00);
        acc01 = WMMA_BF16(fa0, fb1, acc01);
        acc10 = WMMA_BF16(fa1, fb0, acc10);
        acc11 = WMMA_BF16(fa1, fb1, acc11);
    }

    // Epilogue. C/D layout: lanes 0-15 -> M=v, lanes 16-31 -> M=v+8; N=lane&15.
    const int rbase = mt * 32 + ((lane & 16) ? 8 : 0);
#pragma unroll
    for (int t = 0; t < 4; ++t) {
        v8f acc = (t == 0) ? acc00 : (t == 1) ? acc01 : (t == 2) ? acc10 : acc11;
        const int ro = (t >> 1) * 16, co = (t & 1) * 16;
        const int col = nt * 32 + co + rsel;         // 0..2303 = (kk*768 + h*64 + d)
        const float bv = bias[col];
        const int kk  = col / C_;
        const int rem = col - kk * C_;
        const int h = rem >> 6, d = rem & 63;
#pragma unroll
        for (int v = 0; v < 8; ++v) {
            const int row = rbase + ro + v;          // 0..8191
            const int bb = row >> 10, n = row & (N_ - 1);
            const float val = acc[v] + bv;
            if (kk == 0)
                Qb[((size_t)(bb * H_ + h) * N_ + n) * D_ + d] = (bf16)(val * SCALE_);
            else if (kk == 1)
                Kb[((size_t)(bb * H_ + h) * N_ + n) * D_ + d] = (bf16)val;
            else
                Vt[((size_t)(bb * H_ + h) * D_ + d) * N_ + n] = (bf16)val;
        }
    }
}

// ---------------------------------------------------------------------------
// Flash attention: one wave per (head, 16-query tile).  32 keys per step:
// 4 WMMA for S, online softmax (shfl-xor over 16-lane halves), P re-layout
// via per-wave LDS tile, 4 WMMA for O.
// ---------------------------------------------------------------------------
__global__ void __launch_bounds__(128)
attn_kernel(const bf16* __restrict__ Qb, const bf16* __restrict__ Kb,
            const bf16* __restrict__ Vt, bf16* __restrict__ Oo) {
    __shared__ __align__(16) bf16 plds[4][16 * 32];  // 1 KB staging per wave

    const int lane  = threadIdx.x & 31;
    const int wslot = threadIdx.x >> 5;
    const int wave  = blockIdx.x * 4 + wslot;
    const int qt = wave & 63;                        // N/16 = 64 query tiles
    const int hb = wave >> 6;                        // 0..95 = b*H+h
    const int klo  = (lane & 16) ? 8 : 0;
    const int rsel = lane & 15;

    const bf16* Qh = Qb + (size_t)hb * N_ * D_;
    const bf16* Kh = Kb + (size_t)hb * N_ * D_;
    const bf16* Vh = Vt + (size_t)hb * D_ * N_;

    const bf16* qrow = Qh + (size_t)(qt * 16 + rsel) * D_;
    const v16bf aq0 = load_frag(qrow,      klo);     // d 0..31
    const v16bf aq1 = load_frag(qrow + 32, klo);     // d 32..63

    float mrow[8], lrow[8];
#pragma unroll
    for (int v = 0; v < 8; ++v) { mrow[v] = -3.0e38f; lrow[v] = 0.0f; }
    v8f o0 = {}, o1 = {}, o2 = {}, o3 = {};

    bf16* pl = plds[wslot];
    const int rb = (lane & 16) ? 8 : 0;

    for (int m0 = 0; m0 < N_; m0 += 32) {
        // ---- S = Q K^T, 16x32 tile as two 16x16 accs
        const bf16* krow0 = Kh + (size_t)(m0 + rsel) * D_;
        const bf16* krow1 = krow0 + (size_t)16 * D_;
        v8f s0 = {}, s1 = {};
        s0 = WMMA_BF16(aq0, load_frag(krow0,      klo), s0);
        s0 = WMMA_BF16(aq1, load_frag(krow0 + 32, klo), s0);
        s1 = WMMA_BF16(aq0, load_frag(krow1,      klo), s1);
        s1 = WMMA_BF16(aq1, load_frag(krow1 + 32, klo), s1);

        // ---- online softmax per row slot (row = slot v within lane-half)
#pragma unroll
        for (int v = 0; v < 8; ++v) {
            float t = fmaxf(s0[v], s1[v]);
#pragma unroll
            for (int off = 8; off; off >>= 1) t = fmaxf(t, __shfl_xor(t, off, 32));
            const float mn = fmaxf(mrow[v], t);
            const float sc = __expf(mrow[v] - mn);
            const float p0 = __expf(s0[v] - mn);
            const float p1 = __expf(s1[v] - mn);
            float rs = p0 + p1;
#pragma unroll
            for (int off = 8; off; off >>= 1) rs += __shfl_xor(rs, off, 32);
            lrow[v] = lrow[v] * sc + rs;
            mrow[v] = mn;
            o0[v] *= sc; o1[v] *= sc; o2[v] *= sc; o3[v] *= sc;
            s0[v] = p0;  s1[v] = p1;
        }

        // ---- re-layout P: C/D layout -> A layout via LDS round trip
#pragma unroll
        for (int v = 0; v < 8; ++v) {
            pl[(rb + v) * 32 + rsel]      = (bf16)s0[v];
            pl[(rb + v) * 32 + rsel + 16] = (bf16)s1[v];
        }
        asm volatile("s_wait_dscnt 0" ::: "memory");
        const v16bf ap = load_frag(pl + rsel * 32, klo);
        asm volatile("" ::: "memory");  // keep next iter's stores after these loads

        // ---- O += P V  (Vt rows are key-contiguous => clean B frags)
        o0 = WMMA_BF16(ap, load_frag(Vh + (size_t)( 0 + rsel) * N_ + m0, klo), o0);
        o1 = WMMA_BF16(ap, load_frag(Vh + (size_t)(16 + rsel) * N_ + m0, klo), o1);
        o2 = WMMA_BF16(ap, load_frag(Vh + (size_t)(32 + rsel) * N_ + m0, klo), o2);
        o3 = WMMA_BF16(ap, load_frag(Vh + (size_t)(48 + rsel) * N_ + m0, klo), o3);
    }

    // ---- finalize and scatter to O[b][n][h*64 + d] (bf16 for proj GEMM)
    const int bb = hb / H_, h = hb - bb * H_;
    bf16* obase = Oo + (size_t)(bb * N_ + qt * 16 + rb) * C_ + h * D_ + rsel;
#pragma unroll
    for (int v = 0; v < 8; ++v) {
        const float inv = 1.0f / lrow[v];
        bf16* po = obase + (size_t)v * C_;
        po[0]  = (bf16)(o0[v] * inv);
        po[16] = (bf16)(o1[v] * inv);
        po[32] = (bf16)(o2[v] * inv);
        po[48] = (bf16)(o3[v] * inv);
    }
}

// ---------------------------------------------------------------------------
// Proj GEMM: O[8192][768] x Wt[768][768]^T + bias -> out fp32
// ---------------------------------------------------------------------------
__global__ void __launch_bounds__(128)
proj_gemm(const bf16* __restrict__ A, const bf16* __restrict__ Wt,
          const float* __restrict__ bias, float* __restrict__ out) {
    const int lane = threadIdx.x & 31;
    const int wave = blockIdx.x * (blockDim.x >> 5) + (threadIdx.x >> 5);
    const int NT = C_ / 32;                 // 24
    const int mt = wave / NT;
    const int nt = wave - mt * NT;
    const int klo  = (lane & 16) ? 8 : 0;
    const int rsel = lane & 15;

    const bf16* a0 = A  + (size_t)(mt * 32 + rsel) * C_;
    const bf16* a1 = a0 + (size_t)16 * C_;
    const bf16* b0 = Wt + (size_t)(nt * 32 + rsel) * C_;
    const bf16* b1 = b0 + (size_t)16 * C_;

    v8f acc00 = {}, acc01 = {}, acc10 = {}, acc11 = {};
#pragma unroll 4
    for (int k0 = 0; k0 < C_; k0 += 32) {
        v16bf fa0 = load_frag(a0 + k0, klo);
        v16bf fa1 = load_frag(a1 + k0, klo);
        v16bf fb0 = load_frag(b0 + k0, klo);
        v16bf fb1 = load_frag(b1 + k0, klo);
        acc00 = WMMA_BF16(fa0, fb0, acc00);
        acc01 = WMMA_BF16(fa0, fb1, acc01);
        acc10 = WMMA_BF16(fa1, fb0, acc10);
        acc11 = WMMA_BF16(fa1, fb1, acc11);
    }

    const int rbase = mt * 32 + ((lane & 16) ? 8 : 0);
#pragma unroll
    for (int t = 0; t < 4; ++t) {
        v8f acc = (t == 0) ? acc00 : (t == 1) ? acc01 : (t == 2) ? acc10 : acc11;
        const int ro = (t >> 1) * 16, co = (t & 1) * 16;
        const int col = nt * 32 + co + rsel;
        const float bv = bias[col];
#pragma unroll
        for (int v = 0; v < 8; ++v) {
            const int row = rbase + ro + v;
            out[(size_t)row * C_ + col] = acc[v] + bv;
        }
    }
}

// ---------------------------------------------------------------------------
extern "C" void kernel_launch(void* const* d_in, const int* in_sizes, int n_in,
                              void* d_out, int out_size, void* d_ws, size_t ws_size,
                              hipStream_t stream) {
    (void)in_sizes; (void)n_in; (void)out_size; (void)ws_size;
    const float* x     = (const float*)d_in[0];
    const float* Wqkv  = (const float*)d_in[1];
    const float* bqkv  = (const float*)d_in[2];
    const float* Wproj = (const float*)d_in[3];
    const float* bproj = (const float*)d_in[4];
    float* out = (float*)d_out;

    char* ws = (char*)d_ws;
    size_t off = 0;
    auto salloc = [&](size_t bytes) -> void* {
        void* p = ws + off;
        off += (bytes + 255) & ~(size_t)255;
        return p;
    };
    bf16* xb  = (bf16*)salloc((size_t)M_ * C_ * 2);          // 12.6 MB
    bf16* wqt = (bf16*)salloc((size_t)3 * C_ * C_ * 2);      //  3.5 MB
    bf16* wpt = (bf16*)salloc((size_t)C_ * C_ * 2);          //  1.2 MB
    bf16* Qb  = (bf16*)salloc((size_t)M_ * C_ * 2);          // 12.6 MB
    bf16* Kb  = (bf16*)salloc((size_t)M_ * C_ * 2);
    bf16* Vtb = (bf16*)salloc((size_t)M_ * C_ * 2);
    bf16* Ob  = (bf16*)salloc((size_t)M_ * C_ * 2);

    const int nx = M_ * C_;                                   // 6291456
    cvt_f32_bf16<<<(nx + 255) / 256, 256, 0, stream>>>(x, xb, nx);
    transpose_cvt<<<(C_ * 3 * C_ + 255) / 256, 256, 0, stream>>>(Wqkv, wqt, C_, 3 * C_);
    transpose_cvt<<<(C_ * C_ + 255) / 256, 256, 0, stream>>>(Wproj, wpt, C_, C_);

    // 256 * 72 tiles, 4 waves/block
    qkv_gemm<<<(256 * 72) / 4, 128, 0, stream>>>(xb, wqt, bqkv, Qb, Kb, Vtb);
    // 96 heads * 64 query tiles = 6144 waves
    attn_kernel<<<6144 / 4, 128, 0, stream>>>(Qb, Kb, Vtb, Ob);
    // 256 * 24 tiles
    proj_gemm<<<(256 * 24) / 4, 128, 0, stream>>>(Ob, wpt, bproj, out);
}
